// SpatialLSTM_40587440947683
// MI455X (gfx1250) — matrix operand order repaired and checked by reading, not compiled, not run on hardware
//
#include <hip/hip_runtime.h>
#include <hip/hip_bf16.h>

// ---------------------------------------------------------------------------
// SpatialLSTM on MI455X (gfx1250): wavefront over anti-diagonals.
// - Gate/head GEMMs in bf16 via v_wmma_f32_16x16x32_bf16 (f32 accumulate)
// - Weights pre-converted AND pre-transposed (N-major) so the B tile is a
//   pure 16B-chunk copy -> staged with global_load_async_to_lds_b128
//   (ASYNCcnt path), overlapped with the A-tile fp32->bf16 gather.
// ---------------------------------------------------------------------------

#define DIM_D 256
#define DIM_H 256
#define DIM_B 64
#define DIM_M 32
#define DIM_N 32
#define HIN   (DIM_D + 2 * DIM_H)   // 768
#define GOUT  (5 * DIM_H)           // 1280
#define LN_EPS 1e-6f

typedef __attribute__((ext_vector_type(16))) __bf16 v16bf;
typedef __attribute__((ext_vector_type(8)))  __bf16 v8bf;
typedef __attribute__((ext_vector_type(8)))  float  v8f;

union AFragU { v16bf v; v8bf h[2]; };

static __device__ __forceinline__ unsigned int f2bf(float f) {
    unsigned int u = __float_as_uint(f);
    unsigned int r = u + 0x7FFFu + ((u >> 16) & 1u);   // round-to-nearest-even
    return r >> 16;
}

static __device__ __forceinline__ float sigf(float x) {
    return 1.0f / (1.0f + __expf(-x));
}

// ---------------------------------------------------------------------------
// fp32 -> bf16 conversion with transpose: src is (K x Nout) row-major,
// dst is (Nout x K) row-major (N-major), so GEMM B-tiles become contiguous.
// ---------------------------------------------------------------------------
__global__ void slstm_cvt_bf16_t(const float* __restrict__ src,
                                 unsigned short* __restrict__ dst,
                                 int K, int Nout) {
    int i = blockIdx.x * blockDim.x + threadIdx.x;
    if (i < K * Nout) {
        int k = i / Nout;
        int n = i - k * Nout;
        dst[(size_t)n * K + k] = (unsigned short)f2bf(src[i]);
    }
}

// ---------------------------------------------------------------------------
// Unified WMMA GEMM for one anti-diagonal.
// mode 0: gates  A = [x | h_left | h_up] (64 x 768), Wt = W_gate^T bf16,
//                out -> gbuf[cell][b][5H]
// mode 1: head   A = h_new (64 x 256) from hbuf[cur][i], Wt = W_head^T bf16,
//                out -> logits d_out[b][i][j][:]
// Block: 256 threads (8 waves). Each block computes a 64 x 128 output slice:
//   wave w owns N-tile w (16 cols), M-tiles 0..3 (64 rows). K stepped by 32.
// ---------------------------------------------------------------------------
__global__ void slstm_wmma_gemm(int mode, int d, int i_min,
                                const float* __restrict__ x,
                                const float* __restrict__ h_init,
                                const float* __restrict__ hbuf,   // [2][M][B][H]
                                const unsigned short* __restrict__ Wt, // Nout x K
                                const float* __restrict__ bias,   // Nout
                                float* __restrict__ gout,         // gates dest
                                float* __restrict__ lout,         // logits dest
                                int K, int Nout) {
    const int cell = blockIdx.x;
    const int i = i_min + cell;
    const int j = d - i;
    const int p   = (d ^ 1) & 1;   // parity of previous diagonal
    const int cur = d & 1;
    const int nb0 = blockIdx.y * 128;

    const int tid  = threadIdx.x;
    const int wave = tid >> 5;
    const int lane = tid & 31;
    const int laneLo = lane & 15;
    const int hiSel  = lane >> 4;

    __shared__ __align__(64) unsigned short lA[64 * 40];   // 64x32 bf16, row stride 40
    __shared__ __align__(64) unsigned short lB[128 * 48];  // 128x32 bf16 (N-major), stride 48

    // Per-thread A staging assignment: one 8-element chunk of the 64x32 tile.
    const int arow = tid >> 2;          // batch row 0..63
    const int akb  = (tid & 3) * 8;     // k-chunk within step
    // Per-thread B staging: two 16B chunks of the 128x32 tile (512 chunks).
    const unsigned bLds0 = (unsigned)(size_t)&lB[(tid >> 2) * 48 + (tid & 3) * 8];
    const unsigned bLds1 = (unsigned)(size_t)&lB[((tid + 256) >> 2) * 48 + (tid & 3) * 8];

    // Accumulators initialized with bias (broadcast along rows)
    const float bv = bias[nb0 + wave * 16 + laneLo];
    v8f acc[4];
#pragma unroll
    for (int m = 0; m < 4; ++m)
#pragma unroll
        for (int r = 0; r < 8; ++r) acc[m][r] = bv;

    const int ksteps = K >> 5;
    for (int ks = 0; ks < ksteps; ++ks) {
        const int k0 = ks << 5;
        __syncthreads();   // previous iteration's fragment reads complete

        // ---- B tile: async copy global -> LDS, 16B per op, no VGPR data ----
        {
            const unsigned short* g0 = &Wt[(size_t)(nb0 + (tid >> 2)) * K + k0 + (tid & 3) * 8];
            const unsigned short* g1 = &Wt[(size_t)(nb0 + ((tid + 256) >> 2)) * K + k0 + (tid & 3) * 8];
            asm volatile("global_load_async_to_lds_b128 %0, %1, off"
                         :: "v"(bLds0), "v"(g0) : "memory");
            asm volatile("global_load_async_to_lds_b128 %0, %1, off"
                         :: "v"(bLds1), "v"(g1) : "memory");
        }

        // ---- A tile: 8 contiguous fp32 per thread -> bf16 -> one b128 store ----
        {
            const int kg = k0 + akb;
            const float* src;
            if (mode == 1) {
                src = &hbuf[(((size_t)cur * DIM_M + i) * DIM_B + arow) * DIM_H + kg];
            } else if (kg < DIM_D) {
                src = &x[((((size_t)arow * DIM_M + i) * DIM_N + j) * DIM_D) + kg];
            } else if (kg < DIM_D + DIM_H) {
                const int hh = kg - DIM_D;
                src = (j == 0) ? &h_init[hh]
                    : &hbuf[(((size_t)p * DIM_M + i) * DIM_B + arow) * DIM_H + hh];
            } else {
                const int hh = kg - (DIM_D + DIM_H);
                src = (i == 0) ? &h_init[hh]
                    : &hbuf[(((size_t)p * DIM_M + (i - 1)) * DIM_B + arow) * DIM_H + hh];
            }
            const float4 f0 = *reinterpret_cast<const float4*>(src);
            const float4 f1 = *reinterpret_cast<const float4*>(src + 4);
            uint4 pk;
            pk.x = f2bf(f0.x) | (f2bf(f0.y) << 16);
            pk.y = f2bf(f0.z) | (f2bf(f0.w) << 16);
            pk.z = f2bf(f1.x) | (f2bf(f1.y) << 16);
            pk.w = f2bf(f1.z) | (f2bf(f1.w) << 16);
            *reinterpret_cast<uint4*>(&lA[arow * 40 + akb]) = pk;
        }

        // prefetch next K-slab of weights into GL2 (global_prefetch_b8)
        if (ks + 1 < ksteps)
            __builtin_prefetch(&Wt[(size_t)(nb0 + (tid >> 1)) * K + k0 + 32], 0, 1);

        asm volatile("s_wait_asynccnt 0" ::: "memory");   // this wave's B chunks landed
        __syncthreads();                                   // all waves' tiles visible

        // ---- B fragment (32x16, K x N): lane<16 -> col laneLo, K 0..15;
        //      lane>=16 -> col laneLo, K 16..31. 32B-aligned single load. ----
        v16bf bfrag = *reinterpret_cast<const v16bf*>(
            &lB[(wave * 16 + laneLo) * 48 + hiSel * 16]);

        // ---- A fragments (16x32): lane<16 -> row laneLo, K 0..15;
        //      lane>=16 -> row laneLo, K 16..31. ----
#pragma unroll
        for (int m = 0; m < 4; ++m) {
            AFragU af;
            const int rbase = (m * 16 + laneLo) * 40 + hiSel * 16;
            af.h[0] = *reinterpret_cast<const v8bf*>(&lA[rbase]);
            af.h[1] = *reinterpret_cast<const v8bf*>(&lA[rbase + 8]);
            acc[m] = __builtin_amdgcn_wmma_f32_16x16x32_bf16(
                false, af.v, false, bfrag, (short)0, acc[m], false, false);
        }
    }

    // ---- store D: VGPR r, lane l -> row = m*16 + r + 8*hiSel, col = laneLo ----
    const int col = nb0 + wave * 16 + laneLo;
#pragma unroll
    for (int m = 0; m < 4; ++m) {
#pragma unroll
        for (int r = 0; r < 8; ++r) {
            const int row = m * 16 + r + 8 * hiSel;   // batch index
            const float v = acc[m][r];
            if (mode == 0) {
                gout[((size_t)cell * DIM_B + row) * GOUT + col] = v;
            } else {
                lout[((((size_t)row * DIM_M + i) * DIM_N + j) * DIM_D) + col] = v;
            }
        }
    }
}

// ---------------------------------------------------------------------------
// Cell update: gates -> cell state, LayerNorm over H (wave32 shuffle
// reduction), h_new. grid = (cells, 4 batch-groups of 16), block = 256.
// ---------------------------------------------------------------------------
__global__ void slstm_cell_update(int d, int i_min,
                                  const float* __restrict__ g,       // [cell][B][5H]
                                  const float* __restrict__ lnS,
                                  const float* __restrict__ lnB,
                                  float* __restrict__ hbuf,          // [2][M][B][H]
                                  float* __restrict__ cbuf) {        // [2][M][B][H]
    const int cell = blockIdx.x;
    const int i = i_min + cell;
    const int j = d - i;
    const int p   = (d ^ 1) & 1;
    const int cur = d & 1;

    const int wave = threadIdx.x >> 5;
    const int lane = threadIdx.x & 31;

    for (int bb = 0; bb < 2; ++bb) {
        const int b = blockIdx.y * 16 + wave * 2 + bb;
        const float* gc = g + ((size_t)cell * DIM_B + b) * GOUT;

        float cv[8], og[8];
        float sum = 0.0f, ss = 0.0f;
#pragma unroll
        for (int q = 0; q < 8; ++q) {
            const int h = lane + q * 32;
            const float f_c = gc[h];
            const float f_r = gc[DIM_H + h];
            const float i_g = gc[2 * DIM_H + h];
            og[q]           = gc[3 * DIM_H + h];
            const float g_g = gc[4 * DIM_H + h];
            const float cl = (j == 0) ? 0.0f
                : cbuf[(((size_t)p * DIM_M + i) * DIM_B + b) * DIM_H + h];
            const float cu = (i == 0) ? 0.0f
                : cbuf[(((size_t)p * DIM_M + (i - 1)) * DIM_B + b) * DIM_H + h];
            const float c = sigf(f_c) * cl + sigf(f_r) * cu + sigf(i_g) * tanhf(g_g);
            cv[q] = c;
            sum += c;
            ss  += c * c;
        }
        // wave32 reduction over all 256 H values (8 per lane)
#pragma unroll
        for (int o = 16; o > 0; o >>= 1) {
            sum += __shfl_xor(sum, o, 32);
            ss  += __shfl_xor(ss,  o, 32);
        }
        const float mu  = sum * (1.0f / DIM_H);
        const float var = ss * (1.0f / DIM_H) - mu * mu;
        const float rstd = rsqrtf(var + LN_EPS);

#pragma unroll
        for (int q = 0; q < 8; ++q) {
            const int h = lane + q * 32;
            const float ln = (cv[q] - mu) * rstd * lnS[h] + lnB[h];
            const float hn = sigf(og[q]) * tanhf(ln);
            const size_t o = (((size_t)cur * DIM_M + i) * DIM_B + b) * DIM_H + h;
            hbuf[o] = hn;
            cbuf[o] = cv[q];
        }
    }
}

// ---------------------------------------------------------------------------
// Host-side orchestration: 63 diagonals, stream-ordered dependency chain.
// ---------------------------------------------------------------------------
extern "C" void kernel_launch(void* const* d_in, const int* in_sizes, int n_in,
                              void* d_out, int out_size, void* d_ws, size_t ws_size,
                              hipStream_t stream) {
    (void)in_sizes; (void)n_in; (void)out_size; (void)ws_size;

    const float* x        = (const float*)d_in[0];
    const float* W_gate   = (const float*)d_in[1];
    const float* b_gate   = (const float*)d_in[2];
    const float* ln_scale = (const float*)d_in[3];
    const float* ln_bias  = (const float*)d_in[4];
    const float* W_head   = (const float*)d_in[5];
    const float* b_head   = (const float*)d_in[6];
    const float* h_init   = (const float*)d_in[7];
    float* out = (float*)d_out;

    // Workspace layout (20 MB total):
    //   [0, 1966080)            W_gate^T bf16 (1280 x 768 ushort)
    //   [1966080, 2097152)      W_head^T bf16 (256 x 256 ushort)
    //   [2097152, +4MB)         hbuf  [2][32][64][256] f32
    //   [+4MB, +8MB)            cbuf  [2][32][64][256] f32
    //   [+8MB, +18MB)         gbuf  [32][64][1280]   f32
    char* ws = (char*)d_ws;
    unsigned short* Wg_t = (unsigned short*)ws;
    unsigned short* Wh_t = (unsigned short*)(ws + (size_t)HIN * GOUT * 2);
    float* hbuf = (float*)(ws + 2097152);
    float* cbuf = hbuf + (size_t)2 * DIM_M * DIM_B * DIM_H;
    float* gbuf = cbuf + (size_t)2 * DIM_M * DIM_B * DIM_H;

    const int nWg = HIN * GOUT;        // 983040
    const int nWh = DIM_H * DIM_D;     // 65536
    slstm_cvt_bf16_t<<<(nWg + 255) / 256, 256, 0, stream>>>(W_gate, Wg_t, HIN, GOUT);
    slstm_cvt_bf16_t<<<(nWh + 255) / 256, 256, 0, stream>>>(W_head, Wh_t, DIM_H, DIM_D);

    for (int d = 0; d < DIM_M + DIM_N - 1; ++d) {
        const int i_min = (d - (DIM_N - 1) > 0) ? d - (DIM_N - 1) : 0;
        const int i_max = (d < DIM_M - 1) ? d : DIM_M - 1;
        const int cells = i_max - i_min + 1;

        dim3 gGate(cells, GOUT / 128);   // 64 x 1280 per cell
        slstm_wmma_gemm<<<gGate, 256, 0, stream>>>(
            0, d, i_min, x, h_init, hbuf, Wg_t, b_gate, gbuf, nullptr, HIN, GOUT);

        dim3 gCell(cells, DIM_B / 16);
        slstm_cell_update<<<gCell, 256, 0, stream>>>(
            d, i_min, gbuf, ln_scale, ln_bias, hbuf, cbuf);

        dim3 gHead(cells, DIM_D / 128);  // 64 x 256 per cell
        slstm_wmma_gemm<<<gHead, 256, 0, stream>>>(
            1, d, i_min, x, h_init, hbuf, Wh_t, b_head, nullptr, out, DIM_H, DIM_D);
    }
}
